// self_attention_35373350649912
// MI455X (gfx1250) — matrix operand order repaired and checked by reading, not compiled, not run on hardware
//
#include <hip/hip_runtime.h>
#include <hip/hip_bf16.h>

#define DEV __device__ __forceinline__

typedef __attribute__((ext_vector_type(16))) _Float16 v16h;
typedef __attribute__((ext_vector_type(8)))  float    v8f;

constexpr int B  = 4;
constexpr int C  = 192;   // input/output channels
constexpr int HD = 96;    // hidden dim for q/k
constexpr int N  = 4096;  // H*W

// ---------------------------------------------------------------------------
// WMMA helpers (CDNA5 gfx1250, wave32, V_WMMA_F32_16X16X32_F16)
// ---------------------------------------------------------------------------
DEV v8f wmma16(v16h a, v16h b, v8f c) {
  // (neg_a, A, neg_b, B, c_mod, C, reuse_a, reuse_b)
  return __builtin_amdgcn_wmma_f32_16x16x32_f16(false, a, false, b, (short)0, c,
                                                false, false);
}

// A fragment: 16x32 (MxK), source row-major [M][K] with K contiguous.
// lane m=lane%16, h=lane/16: elems 0..7 -> K=h*8+i ; elems 8..15 -> K=16+h*8+i
DEV v16h load_a_f16(const _Float16* base, int stride) {
  int lane = threadIdx.x & 31;
  int m = lane & 15, h = lane >> 4;
  const _Float16* p = base + (size_t)m * stride + h * 8;
  v16h a;
#pragma unroll
  for (int i = 0; i < 8; ++i) a[i] = p[i];
#pragma unroll
  for (int i = 0; i < 8; ++i) a[8 + i] = p[16 + i];
  return a;
}

// Same A fragment but converting from fp32 source on the fly.
DEV v16h load_a_f32(const float* base, int stride) {
  int lane = threadIdx.x & 31;
  int m = lane & 15, h = lane >> 4;
  const float* p = base + (size_t)m * stride + h * 8;
  v16h a;
#pragma unroll
  for (int i = 0; i < 8; ++i) a[i] = (_Float16)p[i];
#pragma unroll
  for (int i = 0; i < 8; ++i) a[8 + i] = (_Float16)p[16 + i];
  return a;
}

// B fragment: 32x16 (KxN), source given TRANSPOSED as row-major [N][K]
// (K contiguous). lane n=lane%16, h=lane/16: elems 0..7 -> K=h*8+i ; elems 8..15 -> K=16+h*8+i
DEV v16h load_bt_f16(const _Float16* base, int stride) {
  int lane = threadIdx.x & 31;
  int n = lane & 15, h = lane >> 4;
  const _Float16* p = base + (size_t)n * stride + h * 8;
  v16h b;
#pragma unroll
  for (int i = 0; i < 8; ++i) b[i] = p[i];
#pragma unroll
  for (int i = 0; i < 8; ++i) b[8 + i] = p[16 + i];
  return b;
}

// ---------------------------------------------------------------------------
// K0: fp32 -> f16 weight conversion
// ---------------------------------------------------------------------------
__global__ void cvt_f16_kernel(const float* __restrict__ src,
                               _Float16* __restrict__ dst, int n) {
  int i = blockIdx.x * 256 + threadIdx.x;
  if (i < n) dst[i] = (_Float16)src[i];
}

// ---------------------------------------------------------------------------
// K1: 1x1 conv projection:  Y[d] = sum_c W[d,c]*X[c,n] + bias[d]
//   MODE 0: Y stored [B][N][D]  (f16, D contiguous)   -> qp, kp
//   MODE 1: Y stored [B][D][N]  (f16, N contiguous)   -> vp
// block: 128 threads (4 waves); grid.x = B * N/16
// ---------------------------------------------------------------------------
template <int D, int MODE>
__global__ void __launch_bounds__(128)
proj_kernel(const float* __restrict__ X,
            const _Float16* __restrict__ Wh,
            const float* __restrict__ bias,
            _Float16* __restrict__ Y) {
  int wg = blockIdx.x;
  int b  = wg >> 8;
  int n0 = (wg & 255) * 16;

  __shared__ _Float16 xT[16][C];  // transposed input tile, 6 KB
  for (int idx = threadIdx.x; idx < 16 * C; idx += 128) {
    int c = idx >> 4, pl = idx & 15;
    xT[pl][c] = (_Float16)X[((size_t)b * C + c) * N + n0 + pl];
  }
  __syncthreads();

  int wave = threadIdx.x >> 5;
  int lane = threadIdx.x & 31;
  int lm = lane & 15, lh = lane >> 4;

  v16h aF[C / 32];
#pragma unroll
  for (int kk = 0; kk < C / 32; ++kk)
    aF[kk] = load_a_f16(&xT[0][0] + kk * 32, C);

  for (int dt = wave; dt < D / 16; dt += 4) {
    v8f acc = {};
#pragma unroll
    for (int kk = 0; kk < C / 32; ++kk) {
      v16h bF = load_bt_f16(Wh + (size_t)(dt * 16) * C + kk * 32, C);
      acc = wmma16(aF[kk], bF, acc);
    }
    int d = dt * 16 + lm;
    float bv = bias[d];
#pragma unroll
    for (int r = 0; r < 8; ++r) {
      int n = n0 + r + 8 * lh;
      float val = acc[r] + bv;
      if (MODE == 0)
        Y[((size_t)b * N + n) * D + d] = (_Float16)val;
      else
        Y[((size_t)b * D + d) * N + n] = (_Float16)val;
    }
  }
}

// ---------------------------------------------------------------------------
// K2: fused  e = qp·kp^T  +  row softmax  -> attn (fp32, written once).
// Scores stay in registers; only normalized attn hits memory.
// block: 512 threads (16 waves); each WG: 16 rows x 4096 cols.
// wave w covers cols [w*256, w*256+256) = 16 column tiles.
// B fragments are double-buffered so the next tile's loads are in flight
// while the current tile's 3 WMMAs execute.
// ---------------------------------------------------------------------------
__global__ void __launch_bounds__(512)
qk_softmax_kernel(const _Float16* __restrict__ qp,
                  const _Float16* __restrict__ kp,
                  float* __restrict__ attn) {
  int wg = blockIdx.x;
  int b  = wg >> 8;
  int n0 = (wg & 255) * 16;

  int wave = threadIdx.x >> 5;
  int lane = threadIdx.x & 31;
  int lm = lane & 15, lh = lane >> 4;

  const _Float16* qbase = qp + ((size_t)b * N + n0) * HD;
  v16h aF[3];
#pragma unroll
  for (int kk = 0; kk < 3; ++kk) aF[kk] = load_a_f16(qbase + kk * 32, HD);

  int col0 = wave * 256;
  const _Float16* kbase = kp + ((size_t)b * N + col0) * HD;

  v8f acc[16];
  v16h bcur[3], bnxt[3];
#pragma unroll
  for (int kk = 0; kk < 3; ++kk) bcur[kk] = load_bt_f16(kbase + kk * 32, HD);

#pragma unroll
  for (int t = 0; t < 16; ++t) {
    if (t < 15) {
      const _Float16* kb = kbase + (size_t)(t + 1) * 16 * HD;
#pragma unroll
      for (int kk = 0; kk < 3; ++kk) bnxt[kk] = load_bt_f16(kb + kk * 32, HD);
    }
    v8f a = {};
#pragma unroll
    for (int kk = 0; kk < 3; ++kk) a = wmma16(aF[kk], bcur[kk], a);
    acc[t] = a;
#pragma unroll
    for (int kk = 0; kk < 3; ++kk) bcur[kk] = bnxt[kk];
  }

  // ---- softmax over 4096 cols, rows = n0 .. n0+15 -------------------------
  __shared__ float red[16][16];     // [wave][row]
  __shared__ float rowred[16];      // final per-row max / sum

  // row max: local over tiles, then over 16 lanes of same half, then waves
  float lmax[8];
#pragma unroll
  for (int r = 0; r < 8; ++r) {
    float m = acc[0][r];
#pragma unroll
    for (int t = 1; t < 16; ++t) m = fmaxf(m, acc[t][r]);
#pragma unroll
    for (int s = 1; s < 16; s <<= 1) m = fmaxf(m, __shfl_xor(m, s, 32));
    lmax[r] = m;
  }
  if (lm == 0) {
#pragma unroll
    for (int r = 0; r < 8; ++r) red[wave][r + 8 * lh] = lmax[r];
  }
  __syncthreads();
  if (threadIdx.x < 16) {
    float m = red[0][threadIdx.x];
#pragma unroll
    for (int w = 1; w < 16; ++w) m = fmaxf(m, red[w][threadIdx.x]);
    rowred[threadIdx.x] = m;
  }
  __syncthreads();
  float rmax[8];
#pragma unroll
  for (int r = 0; r < 8; ++r) rmax[r] = rowred[r + 8 * lh];
  __syncthreads();

  // exp + row sum
  float lsum[8];
#pragma unroll
  for (int r = 0; r < 8; ++r) lsum[r] = 0.f;
#pragma unroll
  for (int t = 0; t < 16; ++t) {
#pragma unroll
    for (int r = 0; r < 8; ++r) {
      float e = __expf(acc[t][r] - rmax[r]);
      acc[t][r] = e;
      lsum[r] += e;
    }
  }
#pragma unroll
  for (int r = 0; r < 8; ++r) {
    float s = lsum[r];
#pragma unroll
    for (int sh = 1; sh < 16; sh <<= 1) s += __shfl_xor(s, sh, 32);
    lsum[r] = s;
  }
  if (lm == 0) {
#pragma unroll
    for (int r = 0; r < 8; ++r) red[wave][r + 8 * lh] = lsum[r];
  }
  __syncthreads();
  if (threadIdx.x < 16) {
    float s = 0.f;
#pragma unroll
    for (int w = 0; w < 16; ++w) s += red[w][threadIdx.x];
    rowred[threadIdx.x] = s;
  }
  __syncthreads();
  float rinv[8];
#pragma unroll
  for (int r = 0; r < 8; ++r) rinv[r] = 1.f / rowred[r + 8 * lh];

  // write normalized attention (single pass over attn-sized data)
#pragma unroll
  for (int t = 0; t < 16; ++t) {
#pragma unroll
    for (int r = 0; r < 8; ++r) {
      int row = n0 + r + 8 * lh;
      int col = col0 + t * 16 + lm;
      attn[((size_t)b * N + row) * N + col] = acc[t][r] * rinv[r];
    }
  }
}

// ---------------------------------------------------------------------------
// K3: z[b,n,c] = sum_m attn[b,n,m] * vp[b,c,m]
// block: 128 threads (4 waves), each wave owns 3 c-tiles; grid = B * N/16.
// attn read once from HBM (fp32 -> f16 fragments in-register); vp is L2-hot.
// Loop unrolled 4x for software pipelining; attn stream prefetched ahead
// (global_prefetch_b8).
// ---------------------------------------------------------------------------
__global__ void __launch_bounds__(128)
av_kernel(const float* __restrict__ attn,
          const _Float16* __restrict__ vph,
          float* __restrict__ z) {
  int wg = blockIdx.x;
  int b  = wg >> 8;
  int n0 = (wg & 255) * 16;

  int wave = threadIdx.x >> 5;
  int lane = threadIdx.x & 31;
  int lm = lane & 15, lh = lane >> 4;

  const float* arow = attn + ((size_t)b * N + n0) * N;
  const float* apf  = arow + (size_t)lm * N + lh * 8;  // per-lane A address

  v8f acc[3];
#pragma unroll
  for (int j = 0; j < 3; ++j) acc[j] = (v8f){};

#pragma unroll 4
  for (int kk = 0; kk < N / 32; ++kk) {  // 128 K-steps
    // prefetch the attn stream 4 K-steps (128 cols) ahead
    if (kk + 4 < N / 32) __builtin_prefetch(apf + (kk + 4) * 32, 0, 1);

    v16h aF = load_a_f32(arow + kk * 32, N);
#pragma unroll
    for (int j = 0; j < 3; ++j) {
      int ct = wave + 4 * j;  // c-tile index 0..11
      v16h bF =
          load_bt_f16(vph + ((size_t)b * C + ct * 16) * N + kk * 32, N);
      acc[j] = wmma16(aF, bF, acc[j]);
    }
  }
#pragma unroll
  for (int j = 0; j < 3; ++j) {
    int c = (wave + 4 * j) * 16 + lm;
#pragma unroll
    for (int r = 0; r < 8; ++r) {
      int n = n0 + r + 8 * lh;
      z[((size_t)b * N + n) * C + c] = acc[j][r];
    }
  }
}

// ---------------------------------------------------------------------------
// K4: out[b,o,p] = sum_c Wf[o,c] * (zview[b,c,p]*res[b,c,p]) + bf[o] + res[b,o,p]
// zview is the raw reinterpretation of z [B,N,C] as [B,C,HW].
// block: 128 threads; grid = B * N/16 (p-tiles of 16).
// ---------------------------------------------------------------------------
__global__ void __launch_bounds__(128)
final_kernel(const float* __restrict__ z,
             const float* __restrict__ res,
             const _Float16* __restrict__ Wfh,
             const float* __restrict__ bf,
             float* __restrict__ out) {
  int wg = blockIdx.x;
  int b  = wg >> 8;
  int p0 = (wg & 255) * 16;

  const float* zb = z + (size_t)b * C * N;    // viewed [C][N]
  const float* rb = res + (size_t)b * C * N;

  __shared__ _Float16 gT[16][C];  // gated tile, transposed to [p][c]
  for (int idx = threadIdx.x; idx < 16 * C; idx += 128) {
    int c = idx >> 4, pl = idx & 15;
    size_t o = (size_t)c * N + p0 + pl;
    gT[pl][c] = (_Float16)(zb[o] * rb[o]);
  }
  __syncthreads();

  int wave = threadIdx.x >> 5;
  int lane = threadIdx.x & 31;
  int lm = lane & 15, lh = lane >> 4;

  v16h aF[C / 32];
#pragma unroll
  for (int kk = 0; kk < C / 32; ++kk)
    aF[kk] = load_a_f16(&gT[0][0] + kk * 32, C);

  for (int ot = wave; ot < C / 16; ot += 4) {
    v8f acc = {};
#pragma unroll
    for (int kk = 0; kk < C / 32; ++kk) {
      v16h bF = load_bt_f16(Wfh + (size_t)(ot * 16) * C + kk * 32, C);
      acc = wmma16(aF[kk], bF, acc);
    }
    int o = ot * 16 + lm;
    float bias = bf[o];
#pragma unroll
    for (int r = 0; r < 8; ++r) {
      int p = p0 + r + 8 * lh;
      size_t oi = ((size_t)b * C + o) * N + p;
      out[oi] = acc[r] + bias + res[oi];
    }
  }
}

// ---------------------------------------------------------------------------
// Host launch
// ---------------------------------------------------------------------------
extern "C" void kernel_launch(void* const* d_in, const int* in_sizes, int n_in,
                              void* d_out, int out_size, void* d_ws,
                              size_t ws_size, hipStream_t stream) {
  const float* q  = (const float*)d_in[0];
  const float* k  = (const float*)d_in[1];
  const float* v  = (const float*)d_in[2];
  const float* Wq = (const float*)d_in[3];
  const float* bq = (const float*)d_in[4];
  const float* Wk = (const float*)d_in[5];
  const float* bk = (const float*)d_in[6];
  const float* Wv = (const float*)d_in[7];
  const float* bv = (const float*)d_in[8];
  const float* Wf = (const float*)d_in[9];
  const float* bf = (const float*)d_in[10];

  float* out  = (float*)d_out;                 // [B, C, H, W]
  float* attn = out + (size_t)B * C * N;       // [B, N, N]

  // workspace carve-up
  char* ws = (char*)d_ws;
  _Float16* Wq_h = (_Float16*)ws;  ws += (size_t)HD * C * sizeof(_Float16);
  _Float16* Wk_h = (_Float16*)ws;  ws += (size_t)HD * C * sizeof(_Float16);
  _Float16* Wv_h = (_Float16*)ws;  ws += (size_t)C * C * sizeof(_Float16);
  _Float16* Wf_h = (_Float16*)ws;  ws += (size_t)C * C * sizeof(_Float16);
  _Float16* qp_h = (_Float16*)ws;  ws += (size_t)B * N * HD * sizeof(_Float16);
  _Float16* kp_h = (_Float16*)ws;  ws += (size_t)B * N * HD * sizeof(_Float16);
  _Float16* vp_h = (_Float16*)ws;  ws += (size_t)B * C * N * sizeof(_Float16);
  float*    zbuf = (float*)ws;     // [B, N, C] fp32

  // K0: weight conversion
  cvt_f16_kernel<<<(HD * C + 255) / 256, 256, 0, stream>>>(Wq, Wq_h, HD * C);
  cvt_f16_kernel<<<(HD * C + 255) / 256, 256, 0, stream>>>(Wk, Wk_h, HD * C);
  cvt_f16_kernel<<<(C * C + 255) / 256, 256, 0, stream>>>(Wv, Wv_h, C * C);
  cvt_f16_kernel<<<(C * C + 255) / 256, 256, 0, stream>>>(Wf, Wf_h, C * C);

  dim3 grid(B * (N / 16));  // 1024 workgroups

  // K1: projections
  proj_kernel<HD, 0><<<grid, 128, 0, stream>>>(q, Wq_h, bq, qp_h);
  proj_kernel<HD, 0><<<grid, 128, 0, stream>>>(k, Wk_h, bk, kp_h);
  proj_kernel<C, 1><<<grid, 128, 0, stream>>>(v, Wv_h, bv, vp_h);

  // K2: fused QK^T + softmax -> attn (written once)
  qk_softmax_kernel<<<grid, 512, 0, stream>>>(qp_h, kp_h, attn);

  // K3: z = attn @ vp^T
  av_kernel<<<grid, 128, 0, stream>>>(attn, vp_h, zbuf);

  // K4: out = conv1x1(z * res, Wf, bf) + res   (res == q)
  final_kernel<<<grid, 128, 0, stream>>>(zbuf, q, Wf_h, bf, out);
}
